// QueryInstanceDecoder_12498354831487
// MI455X (gfx1250) — compile-verified, matrix operand order checked by reading, not run on hardware
//
#include <hip/hip_runtime.h>

typedef unsigned short u16;
typedef unsigned int   u32;
typedef __attribute__((ext_vector_type(8)))  float   v8f;
typedef __attribute__((ext_vector_type(16))) __bf16  bf16x16;
typedef __attribute__((ext_vector_type(4)))  u32     u32x4;

#define NPTS 262144
#define DIM  256
#define NQ   128
#define CIN  72

// ---------------- helpers ----------------

union FragBF {
    bf16x16 v;
    u32x4   q[2];
};

__device__ __forceinline__ u16 f2bf(float f) {
    u32 u = __float_as_uint(f);
    u += 0x7FFFu + ((u >> 16) & 1u);        // round-to-nearest-even
    return (u16)(u >> 16);
}

__device__ __forceinline__ float gelu_f(float x) {
    return 0.5f * x * (1.0f + erff(x * 0.7071067811865475f));
}

__device__ __forceinline__ v8f wmma_bf16(const FragBF& a, const FragBF& b, v8f c) {
    return __builtin_amdgcn_wmma_f32_16x16x32_bf16(false, a.v, false, b.v,
                                                   (short)0, c, false, false);
}

__device__ __forceinline__ float blockReduceSum256(float v, float* red) {
    int tid = threadIdx.x;
    red[tid] = v; __syncthreads();
    for (int s = 128; s > 0; s >>= 1) {
        if (tid < s) red[tid] += red[tid + s];
        __syncthreads();
    }
    float r = red[0]; __syncthreads();
    return r;
}

// ---------------- K0: weight fragment prep ----------------
// B-fragment layout (32x16 bf16 per frag, 1KB): lane L (col n=16*nb+L%16, hi=L/16),
// dword r: r<4 -> K = kb*32+hi*8+2r, r>=4 -> K = kb*32+16+hi*8+2(r-4); pair (K,K+1).
__global__ void prep_weights_kernel(const float* __restrict__ w_ip1,
                                    const float* __restrict__ w_ip2,
                                    const float* __restrict__ w_ph1,
                                    const float* __restrict__ w_ph2,
                                    u16* wf1, u16* wf2, u16* wfp1, u16* wfp2) {
    int t = blockIdx.x * 256 + threadIdx.x;          // 432*256 total
    int fg     = t >> 8;
    int within = t & 255;
    int lane = within >> 3, r = within & 7;
    const float* W; u16* out; int kreal, fidx;
    if (fg < 48)       { W = w_ip1; out = wf1;  kreal = 72;  fidx = fg;       }
    else if (fg < 176) { W = w_ip2; out = wf2;  kreal = 256; fidx = fg - 48;  }
    else if (fg < 304) { W = w_ph1; out = wfp1; kreal = 256; fidx = fg - 176; }
    else               { W = w_ph2; out = wfp2; kreal = 256; fidx = fg - 304; }
    int kb = fidx >> 4, nb = fidx & 15;
    int lo = lane & 15, hi = lane >> 4;
    int n = nb * 16 + lo;
    int k = kb * 32 + ((r < 4) ? (hi * 8 + 2 * r) : (16 + hi * 8 + 2 * (r - 4)));
    float v0 = (k     < kreal) ? W[(size_t)k * 256 + n]       : 0.f;
    float v1 = (k + 1 < kreal) ? W[(size_t)(k + 1) * 256 + n] : 0.f;
    u32 dw = (u32)f2bf(v0) | ((u32)f2bf(v1) << 16);
    ((u32*)(out + (size_t)fidx * 512 + lane * 16))[r] = dw;
}

// ---------------- K_q: query head (small, fp32 VALU) ----------------
__global__ void query_head_kernel(const float* __restrict__ qemb,
                                  const float* __restrict__ lng, const float* __restrict__ lnb,
                                  const float* __restrict__ w1, const float* __restrict__ b1,
                                  const float* __restrict__ w2, const float* __restrict__ b2,
                                  float* __restrict__ qe_out, u16* __restrict__ qm_out) {
    __shared__ float row[256], tn[256], gg[256], red[256];
    int q = blockIdx.x, tid = threadIdx.x;
    float x = qemb[q * 256 + tid];
    row[tid] = x;
    float s = blockReduceSum256(x, red);
    float mu = s * (1.f / 256.f);
    float d = x - mu;
    float var = blockReduceSum256(d * d, red);
    float rs = rsqrtf(var * (1.f / 256.f) + 1e-5f);
    tn[tid] = d * rs * lng[tid] + lnb[tid];
    __syncthreads();
    float a = b1[tid];
    for (int k = 0; k < 256; ++k) a += tn[k] * w1[k * 256 + tid];
    gg[tid] = gelu_f(a);
    __syncthreads();
    float a2 = b2[tid];
    for (int k = 0; k < 256; ++k) a2 += gg[k] * w2[k * 256 + tid];
    float qe = row[tid] + a2;
    qe_out[q * 256 + tid] = qe;
    float nn = blockReduceSum256(qe * qe, red);
    float inv = 1.f / fmaxf(sqrtf(nn), 1e-12f);
    qm_out[q * 256 + tid] = f2bf(qe * inv);
}

// ---------------- K1: point pipeline (64 rows / workgroup, 8 waves) ----------------
// LDS: actA u16[64][104] | act1 u16[64][264] | act2 u16[64][264] | F f32[64][257]
#define LDS_ACTA 0
#define LDS_ACT1 13312
#define LDS_ACT2 47104
#define LDS_F    80896
#define K1_LDS   146688

// MODE 0: F = acc+bias ; MODE 1: act2 = bf16(gelu(acc+bias)) ; MODE 2: pe = F+acc+bias -> F & global
template <int MODE>
__device__ __forceinline__ void gemm64(const u16* __restrict__ act, int astride, int kBlocks,
                                       const u16* __restrict__ wfrag,
                                       const float* __restrict__ bias,
                                       float* F, u16* outAct,
                                       float* __restrict__ gOut, int row0) {
    int tid = threadIdx.x;
    int lane = tid & 31, wv = tid >> 5;
    int m = wv >> 1, nbh = wv & 1;
    int lo = lane & 15, hi = lane >> 4;
    v8f acc[8] = {};
#pragma unroll 1
    for (int kb = 0; kb < kBlocks; ++kb) {
        FragBF a;
        const u16* ap = act + (m * 16 + lo) * astride + kb * 32 + hi * 8;
        a.q[0] = *(const u32x4*)ap;
        a.q[1] = *(const u32x4*)(ap + 16);
#pragma unroll
        for (int j = 0; j < 8; ++j) {
            int nb = nbh * 8 + j;
            const u16* bp = wfrag + (size_t)((kb * 16 + nb) << 9) + (lane << 4);
            FragBF b;
            b.q[0] = *(const u32x4*)bp;
            b.q[1] = *(const u32x4*)(bp + 8);
            acc[j] = wmma_bf16(a, b, acc[j]);
        }
    }
#pragma unroll
    for (int j = 0; j < 8; ++j) {
        int col = nbh * 128 + j * 16 + lo;
        float bb = bias[col];
#pragma unroll
        for (int r = 0; r < 8; ++r) {
            int rowT = m * 16 + hi * 8 + r;
            float v = acc[j][r] + bb;
            if (MODE == 0) {
                F[rowT * 257 + col] = v;
            } else if (MODE == 1) {
                outAct[rowT * 264 + col] = f2bf(gelu_f(v));
            } else {
                float pe = F[rowT * 257 + col] + v;
                F[rowT * 257 + col] = pe;
                gOut[(size_t)(row0 + rowT) * 256 + col] = pe;
            }
        }
    }
}

__device__ __forceinline__ void ep_ln_64(const float* __restrict__ F,
                                         const float* __restrict__ g,
                                         const float* __restrict__ b,
                                         bool doGelu, u16* __restrict__ out) {
    int tid = threadIdx.x;
    int row = tid >> 2, part = tid & 3;
    const float* fr = F + row * 257;
    int k0 = part * 64;
    float s = 0.f;
#pragma unroll 8
    for (int k = k0; k < k0 + 64; ++k) s += fr[k];
    s += __shfl_xor(s, 1); s += __shfl_xor(s, 2);
    float mu = s * (1.f / 256.f);
    float var = 0.f;
#pragma unroll 8
    for (int k = k0; k < k0 + 64; ++k) { float d = fr[k] - mu; var += d * d; }
    var += __shfl_xor(var, 1); var += __shfl_xor(var, 2);
    float rs = rsqrtf(var * (1.f / 256.f) + 1e-5f);
    u16* orow = out + row * 264;
#pragma unroll 8
    for (int k = k0; k < k0 + 64; ++k) {
        float t = (fr[k] - mu) * rs * g[k] + b[k];
        if (doGelu) t = gelu_f(t);
        orow[k] = f2bf(t);
    }
}

__global__ void point_pipeline_kernel(const float* __restrict__ pf,
        const u16* __restrict__ wf1, const float* __restrict__ ib1,
        const float* __restrict__ ilng, const float* __restrict__ ilnb,
        const u16* __restrict__ wf2, const float* __restrict__ ib2,
        const float* __restrict__ plng, const float* __restrict__ plnb,
        const u16* __restrict__ wfp1, const float* __restrict__ pb1,
        const u16* __restrict__ wfp2, const float* __restrict__ pb2,
        float* __restrict__ pe_out, float* __restrict__ rnorm) {
    extern __shared__ char smem[];
    u16*   actA = (u16*)(smem + LDS_ACTA);
    u16*   act1 = (u16*)(smem + LDS_ACT1);
    u16*   act2 = (u16*)(smem + LDS_ACT2);
    float* F    = (float*)(smem + LDS_F);
    int tid = threadIdx.x;
    int row0 = blockIdx.x * 64;

#pragma unroll 1
    for (int idx = tid; idx < 64 * 96; idx += 256) {
        int r = idx / 96, c = idx % 96;
        float v = (c < CIN) ? pf[(size_t)(row0 + r) * CIN + c] : 0.f;
        actA[r * 104 + c] = f2bf(v);
    }
    __syncthreads();
    gemm64<0>(actA, 104, 3, wf1, ib1, F, nullptr, nullptr, 0);
    __syncthreads();
    ep_ln_64(F, ilng, ilnb, true, act1);
    __syncthreads();
    gemm64<0>(act1, 264, 8, wf2, ib2, F, nullptr, nullptr, 0);
    __syncthreads();
    ep_ln_64(F, plng, plnb, false, act1);
    __syncthreads();
    gemm64<1>(act1, 264, 8, wfp1, pb1, nullptr, act2, nullptr, 0);
    __syncthreads();
    gemm64<2>(act2, 264, 8, wfp2, pb2, F, nullptr, pe_out, row0);
    __syncthreads();
    // inverse L2 norm per row
    int row = tid >> 2, part = tid & 3;
    const float* fr = F + row * 257;
    float s = 0.f;
#pragma unroll 8
    for (int k = part * 64; k < part * 64 + 64; ++k) { float x = fr[k]; s += x * x; }
    s += __shfl_xor(s, 1); s += __shfl_xor(s, 2);
    if (part == 0) rnorm[row0 + row] = 1.f / fmaxf(sqrtf(s), 1e-12f);
}

// ---------------- K2: similarity logits (Q=128 x 128 pts per WG, acc[8]) ----------------
#define K2_PTS 128
#define K2_LDS (K2_PTS * 264 * 2)
__global__ void sim_kernel(const float* __restrict__ pe, const float* __restrict__ rnorm,
                           const u16* __restrict__ qm, const float* __restrict__ lsc,
                           float* __restrict__ mlout) {
    extern __shared__ char smem[];
    u16* pmT = (u16*)smem;                     // [128 pts][264] bf16, normalized
    int tid = threadIdx.x;
    int row0 = blockIdx.x * K2_PTS;
#pragma unroll 1
    for (int i = 0; i < K2_PTS; ++i) {
        float rn = rnorm[row0 + i];
        pmT[i * 264 + tid] = f2bf(pe[(size_t)(row0 + i) * 256 + tid] * rn);
    }
    __syncthreads();
    int lane = tid & 31, wv = tid >> 5;
    int lo = lane & 15, hi = lane >> 4;
    v8f acc[8] = {};
#pragma unroll 1
    for (int kb = 0; kb < 8; ++kb) {
        FragBF a;
        const u16* ap = qm + (size_t)(wv * 16 + lo) * 256 + kb * 32 + hi * 8;
        a.q[0] = *(const u32x4*)ap;
        a.q[1] = *(const u32x4*)(ap + 16);
#pragma unroll
        for (int nb = 0; nb < 8; ++nb) {
            const u16* bp = pmT + (nb * 16 + lo) * 264 + kb * 32 + hi * 8;
            FragBF b;
            b.q[0] = *(const u32x4*)bp;
            b.q[1] = *(const u32x4*)(bp + 16);
            acc[nb] = wmma_bf16(a, b, acc[nb]);
        }
    }
    float scale = expf(lsc[0]);
#pragma unroll
    for (int nb = 0; nb < 8; ++nb) {
#pragma unroll
        for (int r = 0; r < 8; ++r) {
            int q = wv * 16 + hi * 8 + r;
            int p = row0 + nb * 16 + lo;
            mlout[(size_t)q * NPTS + p] = scale * acc[nb][r];
        }
    }
}

// ---------------- K3: softmax-weighted sum partials (deterministic) ----------------
// 512 threads = 16 waves; wave w: q-block = w%8, d-half = w/8 -> acc[8]
#define NT3 2048
#define NB3 (NPTS / NT3)          // 128 blocks
__global__ void wsum_kernel(const float* __restrict__ ml, const float* __restrict__ pe,
                            float* __restrict__ Rpart, float* __restrict__ Spart) {
    __shared__ u16   Ebf[128 * 40];      // exp(logits) tile [q][32], bf16
    __shared__ u16   peT[256 * 40];      // transposed pe tile [d][32], bf16
    __shared__ float P[512][8];          // per-thread sumexp partials
    int tid = threadIdx.x;
    int lane = tid & 31, wv = tid >> 5;
    int lo = lane & 15, hi = lane >> 4;
    int mb = wv & 7, dh = wv >> 3;
    int n0 = blockIdx.x * NT3;
    float ps[8];
#pragma unroll
    for (int i = 0; i < 8; ++i) ps[i] = 0.f;
    v8f acc[8] = {};
#pragma unroll 1
    for (int ks = 0; ks < NT3 / 32; ++ks) {
        int nb0 = n0 + ks * 32;
        __syncthreads();
#pragma unroll
        for (int i = 0; i < 8; ++i) {               // E: 128q x 32n, 512 threads
            int idx = i * 512 + tid;
            int q = idx >> 5, c = idx & 31;
            float e = expf(ml[(size_t)q * NPTS + nb0 + c]);
            ps[i] += e;
            Ebf[q * 40 + c] = f2bf(e);
        }
#pragma unroll 1
        for (int i = 0; i < 16; ++i) {              // pe transposed: [d][point]
            int p = i * 2 + (tid >> 8);
            int c = tid & 255;
            peT[c * 40 + p] = f2bf(pe[(size_t)(nb0 + p) * 256 + c]);
        }
        __syncthreads();
        FragBF a;
        const u16* ap = Ebf + (mb * 16 + lo) * 40 + hi * 8;
        a.q[0] = *(const u32x4*)ap;
        a.q[1] = *(const u32x4*)(ap + 16);
#pragma unroll
        for (int db = 0; db < 8; ++db) {
            const u16* bp = peT + ((dh * 8 + db) * 16 + lo) * 40 + hi * 8;
            FragBF b;
            b.q[0] = *(const u32x4*)bp;
            b.q[1] = *(const u32x4*)(bp + 16);
            acc[db] = wmma_bf16(a, b, acc[db]);
        }
    }
#pragma unroll
    for (int i = 0; i < 8; ++i) P[tid][i] = ps[i];
    __syncthreads();
    if (tid < 128) {
        int q = tid, g = q & 15, i = q >> 4;
        float s = 0.f;
        for (int l = 0; l < 32; ++l) s += P[g * 32 + l][i];   // fixed order
        Spart[blockIdx.x * 128 + q] = s;
    }
#pragma unroll
    for (int db = 0; db < 8; ++db) {
#pragma unroll
        for (int r = 0; r < 8; ++r) {
            int q = mb * 16 + hi * 8 + r;
            int d = (dh * 8 + db) * 16 + lo;
            Rpart[((size_t)blockIdx.x * 128 + q) * 256 + d] = acc[db][r];
        }
    }
}

// ---------------- K4: refine + score head ----------------
__global__ void final_kernel(const float* __restrict__ Rpart, const float* __restrict__ Spart,
                             const float* __restrict__ qe,
                             const float* __restrict__ lng, const float* __restrict__ lnb,
                             const float* __restrict__ w1, const float* __restrict__ b1,
                             const float* __restrict__ w2, const float* __restrict__ b2,
                             float* __restrict__ refined_out, float* __restrict__ score_out) {
    __shared__ float tn[256], gg[256], red[256];
    int q = blockIdx.x, tid = threadIdx.x;
    float sp = (tid < NB3) ? Spart[tid * 128 + q] : 0.f;
    float S = blockReduceSum256(sp, red);
    float r = 0.f;
#pragma unroll 4
    for (int b = 0; b < NB3; ++b) r += Rpart[((size_t)b * 128 + q) * 256 + tid];
    float ref = qe[q * 256 + tid] + r / S;
    refined_out[q * 256 + tid] = ref;
    float s = blockReduceSum256(ref, red);
    float mu = s * (1.f / 256.f);
    float d = ref - mu;
    float var = blockReduceSum256(d * d, red);
    float rs = rsqrtf(var * (1.f / 256.f) + 1e-5f);
    tn[tid] = d * rs * lng[tid] + lnb[tid];
    __syncthreads();
    float a = b1[tid];
    for (int k = 0; k < 256; ++k) a += tn[k] * w1[k * 256 + tid];
    gg[tid] = gelu_f(a);
    __syncthreads();
    float p = gg[tid] * w2[tid];
    float sc = blockReduceSum256(p, red);
    if (tid == 0) score_out[q] = sc + b2[0];
}

// ---------------- launch ----------------
extern "C" void kernel_launch(void* const* d_in, const int* in_sizes, int n_in,
                              void* d_out, int out_size, void* d_ws, size_t ws_size,
                              hipStream_t stream) {
    (void)in_sizes; (void)n_in; (void)out_size; (void)ws_size;
    const float* pf     = (const float*)d_in[0];
    const float* ip_w1  = (const float*)d_in[1];
    const float* ip_b1  = (const float*)d_in[2];
    const float* ip_lng = (const float*)d_in[3];
    const float* ip_lnb = (const float*)d_in[4];
    const float* ip_w2  = (const float*)d_in[5];
    const float* ip_b2  = (const float*)d_in[6];
    const float* ph_lng = (const float*)d_in[7];
    const float* ph_lnb = (const float*)d_in[8];
    const float* ph_w1  = (const float*)d_in[9];
    const float* ph_b1  = (const float*)d_in[10];
    const float* ph_w2  = (const float*)d_in[11];
    const float* ph_b2  = (const float*)d_in[12];
    const float* q_emb  = (const float*)d_in[13];
    const float* qh_lng = (const float*)d_in[14];
    const float* qh_lnb = (const float*)d_in[15];
    const float* qh_w1  = (const float*)d_in[16];
    const float* qh_b1  = (const float*)d_in[17];
    const float* qh_w2  = (const float*)d_in[18];
    const float* qh_b2  = (const float*)d_in[19];
    const float* sh_lng = (const float*)d_in[20];
    const float* sh_lnb = (const float*)d_in[21];
    const float* sh_w1  = (const float*)d_in[22];
    const float* sh_b1  = (const float*)d_in[23];
    const float* sh_w2  = (const float*)d_in[24];
    const float* sh_b2  = (const float*)d_in[25];
    const float* lsc    = (const float*)d_in[26];

    char* ws = (char*)d_ws;
    u16*   wf1   = (u16*)(ws + 0);            // 48 frags
    u16*   wf2   = (u16*)(ws + 49152);        // 128 frags
    u16*   wfp1  = (u16*)(ws + 180224);
    u16*   wfp2  = (u16*)(ws + 311296);
    float* rnorm = (float*)(ws + 442368);     // [N]
    u16*   qm    = (u16*)(ws + 1490944);      // [128][256] bf16
    float* qe    = (float*)(ws + 1556480);    // [128][256]
    float* Rpart = (float*)(ws + 1687552);    // [128][128][256] f32
    float* Spart = (float*)(ws + 18464768);   // [128][128]

    float* out         = (float*)d_out;
    float* mask_logits = out;                                   // [128][262144]
    float* score       = out + (size_t)NQ * NPTS;               // [128]
    float* pe_out      = score + NQ;                            // [N][256]
    float* refined     = pe_out + (size_t)NPTS * DIM;           // [128][256]

    hipLaunchKernelGGL(prep_weights_kernel, dim3(432), dim3(256), 0, stream,
                       ip_w1, ip_w2, ph_w1, ph_w2, wf1, wf2, wfp1, wfp2);
    hipLaunchKernelGGL(query_head_kernel, dim3(NQ), dim3(256), 0, stream,
                       q_emb, qh_lng, qh_lnb, qh_w1, qh_b1, qh_w2, qh_b2, qe, qm);
    hipLaunchKernelGGL(point_pipeline_kernel, dim3(NPTS / 64), dim3(256), K1_LDS, stream,
                       pf, wf1, ip_b1, ip_lng, ip_lnb, wf2, ip_b2,
                       ph_lng, ph_lnb, wfp1, ph_b1, wfp2, ph_b2, pe_out, rnorm);
    hipLaunchKernelGGL(sim_kernel, dim3(NPTS / K2_PTS), dim3(256), K2_LDS, stream,
                       pe_out, rnorm, qm, lsc, mask_logits);
    hipLaunchKernelGGL(wsum_kernel, dim3(NB3), dim3(512), 0, stream,
                       mask_logits, pe_out, Rpart, Spart);
    hipLaunchKernelGGL(final_kernel, dim3(NQ), dim3(256), 0, stream,
                       Rpart, Spart, qe, sh_lng, sh_lnb, sh_w1, sh_b1, sh_w2, sh_b2,
                       refined, score);
}
